// da_47132971106554
// MI455X (gfx1250) — matrix operand order repaired and checked by reading, not compiled, not run on hardware
//
#include <hip/hip_runtime.h>
#include <hip/hip_bf16.h>
#include <math.h>

// ---------- types ----------
typedef __attribute__((ext_vector_type(16))) __bf16 bf16x16;
typedef __attribute__((ext_vector_type(8)))  float  floatx8;

struct __align__(16) bf8 { __bf16 h[8]; };

// ---------- problem constants ----------
#define BB   2
#define CH   512
#define C1   256
#define HH   64
#define WW   64
#define HW   4096          // 64*64
#define NN   8192          // BB*HW
#define KDC  4608          // 512*9
#define EPSN 1e-5f

static __device__ __forceinline__ float sigm(float v) {
    return 1.0f / (1.0f + __expf(-v));
}

// =====================================================================
// f32 -> bf16 convert
// =====================================================================
__global__ void k_f32_to_bf16(const float* __restrict__ in, __bf16* __restrict__ out, int n) {
    int i = blockIdx.x * 256 + threadIdx.x;
    if (i < n) out[i] = (__bf16)in[i];
}

// =====================================================================
// depthwise 3x3, pad 1, on x[:, :256]; writes bf16 [c(256)][n(8192)]
// =====================================================================
__global__ void k_dwconv(const float* __restrict__ x, const float* __restrict__ w,
                         const float* __restrict__ b, __bf16* __restrict__ out) {
    int idx = blockIdx.x * 256 + threadIdx.x;        // < 256*8192
    int c = idx >> 13, n = idx & 8191;
    int bb = n >> 12, hw = n & 4095, y = hw >> 6, xc = hw & 63;
    const float* xp = x + ((size_t)bb * CH + c) * HW;
    float acc = b[c];
#pragma unroll
    for (int ky = 0; ky < 3; ++ky) {
#pragma unroll
        for (int kx = 0; kx < 3; ++kx) {
            int yy = y + ky - 1, xx = xc + kx - 1;
            if (yy >= 0 && yy < HH && xx >= 0 && xx < WW)
                acc += xp[yy * WW + xx] * w[c * 9 + ky * 3 + kx];
        }
    }
    out[idx] = (__bf16)acc;
}

// =====================================================================
// bf16 WMMA GEMM: D = A(MxK) * B(KxN) + bias.
// Block tile 128x128, BK=32; 8 waves, each owns 64x32 (4 M-tiles x 2 N-tiles
// = 8 WMMA per K-step with A-fragment reuse across the 2 N-tiles).
// A tile staged with GLOBAL_LOAD_ASYNC_TO_LDS_B128 (ASYNCcnt path);
// B tile staged through VGPRs (needs [n][k] transpose for the B layout).
// =====================================================================
__global__ void k_gemm_bf16(const __bf16* __restrict__ A, const __bf16* __restrict__ Bm,
                            const float* __restrict__ bias,
                            float* __restrict__ out_cn, float* __restrict__ out_nc,
                            int M, int N, int K) {
    __shared__ __align__(16) __bf16 As[128][40];   // [m][k], row stride 80B (16B aligned)
    __shared__ __align__(16) __bf16 Bs[128][40];   // [n][k] transposed

    const int tid  = threadIdx.x;
    const int m0   = blockIdx.y * 128;
    const int n0   = blockIdx.x * 128;
    const int w    = tid >> 5, lane = tid & 31;
    const int wm   = (w & 1) * 64;                 // wave M base within block tile
    const int wn   = (w >> 1) * 32;                // wave N base within block tile

    floatx8 acc[4][2] = {};

    const int la = lane & 15;                      // row/col within 16
    const int kg = lane >> 4;                      // K-group select
    const int ka = kg * 8;                         // A fragment K sub-block
    const int kb = kg * 16;                        // unused (B fragments also use ka)

    for (int k0 = 0; k0 < K; k0 += 32) {
        // ---- A tile 128x32: async global->LDS, 2 x b128 per thread ----
#pragma unroll
        for (int i = 0; i < 2; ++i) {
            int li  = tid + i * 256;
            int row = li >> 2;
            int co  = (li & 3) * 8;
            unsigned lds  = (unsigned)(size_t)&As[row][co];   // low 32 bits = LDS offset
            const __bf16* gp = &A[(size_t)(m0 + row) * K + k0 + co];
            asm volatile("global_load_async_to_lds_b128 %0, %1, off"
                         :: "v"(lds), "v"(gp) : "memory");
        }
        // ---- B tile 32x128 -> transposed Bs[n][k], 2 x b128 per thread ----
#pragma unroll
        for (int i = 0; i < 2; ++i) {
            int li  = tid + i * 256;
            int row = li >> 4;                     // k local 0..31
            int co  = (li & 15) * 8;               // n local
            bf8 v = *(const bf8*)&Bm[(size_t)(k0 + row) * N + n0 + co];
#pragma unroll
            for (int j = 0; j < 8; ++j) Bs[co + j][row] = v.h[j];
        }
        if (k0 + 32 < K)   // speculative prefetch of next B tile -> global_prefetch_b8
            __builtin_prefetch((const void*)&Bm[(size_t)(k0 + 32 + (tid >> 4)) * N + n0], 0, 1);
        asm volatile("s_wait_asynccnt 0x0" ::: "memory");
        __syncthreads();

        // ---- B fragments for the wave's two N-tiles ----
        bf16x16 bfr[2];
#pragma unroll
        for (int nt = 0; nt < 2; ++nt) {
            int colr = wn + nt * 16 + la;
            bf8 b0 = *(const bf8*)&Bs[colr][ka];
            bf8 b1 = *(const bf8*)&Bs[colr][16 + ka];
#pragma unroll
            for (int i = 0; i < 8; ++i) { bfr[nt][i] = b0.h[i]; bfr[nt][8 + i] = b1.h[i]; }
        }
        // ---- 4 A fragments x 2 N-tiles = 8 WMMA ----
#pragma unroll
        for (int j = 0; j < 4; ++j) {
            int row = wm + j * 16 + la;
            bf8 a0 = *(const bf8*)&As[row][ka];
            bf8 a1 = *(const bf8*)&As[row][16 + ka];
            bf16x16 af;
#pragma unroll
            for (int i = 0; i < 8; ++i) { af[i] = a0.h[i]; af[8 + i] = a1.h[i]; }
#pragma unroll
            for (int nt = 0; nt < 2; ++nt)
                acc[j][nt] = __builtin_amdgcn_wmma_f32_16x16x32_bf16(
                                 false, af, false, bfr[nt], (short)0, acc[j][nt], false, false);
        }
        __syncthreads();
    }

    // epilogue: C/D layout -> VGPR r holds M = r + 8*(lane>>4), N = lane&15
#pragma unroll
    for (int j = 0; j < 4; ++j) {
#pragma unroll
        for (int nt = 0; nt < 2; ++nt) {
#pragma unroll
            for (int r = 0; r < 8; ++r) {
                int m = m0 + wm + j * 16 + r + (kg << 3);
                int n = n0 + wn + nt * 16 + la;
                float v = acc[j][nt][r] + (bias ? bias[m] : 0.0f);
                if (out_cn) out_cn[(size_t)m * N + n] = v;
                if (out_nc) out_nc[(size_t)n * M + m] = v;
            }
        }
    }
}

// =====================================================================
// instance-norm stats per (b,c) over HW
// =====================================================================
__global__ void k_inst_stats(const float* __restrict__ d, float* __restrict__ mean,
                             float* __restrict__ rstd) {
    int bc = blockIdx.x;                  // b*512 + c
    int b = bc >> 9, c = bc & 511;
    const float* p = d + (size_t)c * NN + (size_t)b * HW;
    float s = 0.f, s2 = 0.f;
    for (int i = threadIdx.x; i < HW; i += 256) { float v = p[i]; s += v; s2 += v * v; }
    __shared__ float sh[256], sh2[256];
    sh[threadIdx.x] = s; sh2[threadIdx.x] = s2; __syncthreads();
    for (int o = 128; o > 0; o >>= 1) {
        if (threadIdx.x < o) { sh[threadIdx.x] += sh[threadIdx.x + o]; sh2[threadIdx.x] += sh2[threadIdx.x + o]; }
        __syncthreads();
    }
    if (threadIdx.x == 0) {
        float mu = sh[0] * (1.0f / HW);
        float var = sh2[0] * (1.0f / HW) - mu * mu;
        mean[bc] = mu; rstd[bc] = rsqrtf(var + EPSN);
    }
}

__global__ void k_inst_norm(float* __restrict__ d, const float* __restrict__ mean,
                            const float* __restrict__ rstd) {
    int idx = blockIdx.x * 256 + threadIdx.x;   // < 512*8192
    int c = idx >> 13, n = idx & 8191, b = n >> 12;
    int bc = (b << 9) | c;
    d[idx] = (d[idx] - mean[bc]) * rstd[bc];
}

// =====================================================================
// offset conv: 3x3, 512 -> 18, pad 1; out layout (b,18,hw)
// =====================================================================
__global__ void k_offconv(const float* __restrict__ d, const float* __restrict__ w,
                          const float* __restrict__ bia, float* __restrict__ out) {
    int idx = blockIdx.x * 256 + threadIdx.x;   // < 18*8192
    int oc = idx >> 13, n = idx & 8191;
    int b = n >> 12, hw = n & 4095, y = hw >> 6, xc = hw & 63;
    float acc = bia[oc];
    for (int ic = 0; ic < CH; ++ic) {
        const float* dp = d + (size_t)ic * NN + (size_t)b * HW;
        const float* wp = w + (size_t)oc * KDC + ic * 9;
#pragma unroll
        for (int ky = 0; ky < 3; ++ky) {
#pragma unroll
            for (int kx = 0; kx < 3; ++kx) {
                int yy = y + ky - 1, xx = xc + kx - 1;
                if (yy >= 0 && yy < HH && xx >= 0 && xx < WW)
                    acc += dp[yy * WW + xx] * wp[ky * 3 + kx];
            }
        }
    }
    out[((size_t)b * 18 + oc) * HW + hw] = acc;
}

// =====================================================================
// deformable bilinear sampling -> bf16 K-major matrix [c*9+t][n]
// =====================================================================
__global__ void k_sample(const float* __restrict__ d, const float* __restrict__ off,
                         __bf16* __restrict__ samp) {
    int idx = blockIdx.x * 256 + threadIdx.x;   // < 4608*8192
    int n = idx & 8191, kRow = idx >> 13;
    int c = kRow / 9, t = kRow - c * 9;
    int b = n >> 12, hw = n & 4095, y = hw >> 6, xc = hw & 63;

    float oy = off[((size_t)b * 18 + 2 * t    ) * HW + hw];
    float ox = off[((size_t)b * 18 + 2 * t + 1) * HW + hw];
    float py = (float)y + (float)(t / 3 - 1) + oy;
    float px = (float)xc + (float)(t % 3 - 1) + ox;
    float fy = floorf(py), fx = floorf(px);
    float ly = py - fy, lx = px - fx;
    int y0 = (int)fy, x0 = (int)fx;

    const float* dp = d + (size_t)c * NN + (size_t)b * HW;
    float v = 0.f;
#pragma unroll
    for (int dy = 0; dy < 2; ++dy) {
#pragma unroll
        for (int dx = 0; dx < 2; ++dx) {
            int yy = y0 + dy, xx = x0 + dx;
            float wgt = (dy ? ly : 1.f - ly) * (dx ? lx : 1.f - lx);
            if (yy >= 0 && yy < HH && xx >= 0 && xx < WW)
                v += wgt * dp[yy * WW + xx];
        }
    }
    samp[idx] = (__bf16)v;
}

// =====================================================================
// x2 per-channel max / mean over HW
// =====================================================================
__global__ void k_x2_stats(const float* __restrict__ x, float* __restrict__ xmax,
                           float* __restrict__ xavg) {
    int blk = blockIdx.x;                     // < 512 : b*256 + c2
    int b = blk >> 8, c2 = blk & 255;
    const float* p = x + ((size_t)b * CH + C1 + c2) * HW;
    float mx = -INFINITY, s = 0.f;
    for (int i = threadIdx.x; i < HW; i += 256) { float v = p[i]; mx = fmaxf(mx, v); s += v; }
    __shared__ float shm[256], shs[256];
    shm[threadIdx.x] = mx; shs[threadIdx.x] = s; __syncthreads();
    for (int o = 128; o > 0; o >>= 1) {
        if (threadIdx.x < o) { shm[threadIdx.x] = fmaxf(shm[threadIdx.x], shm[threadIdx.x + o]);
                               shs[threadIdx.x] += shs[threadIdx.x + o]; }
        __syncthreads();
    }
    if (threadIdx.x == 0) { xmax[blk] = shm[0]; xavg[blk] = shs[0] * (1.0f / HW); }
}

// =====================================================================
// gate g = sigmoid(dsc2(xmax)+dsc2(xavg)); 1x1 spatial => only dw center tap
// =====================================================================
__global__ void k_gate(const float* __restrict__ xmax, const float* __restrict__ xavg,
                       const float* __restrict__ adw_w, const float* __restrict__ adw_b,
                       const float* __restrict__ apw_w, const float* __restrict__ apw_b,
                       float* __restrict__ g) {
    int idx = blockIdx.x * 256 + threadIdx.x;  // < 1024 : b*512 + o
    if (idx >= BB * CH) return;
    int b = idx >> 9, o = idx & 511;
    float acc = 2.0f * apw_b[o];
    for (int c = 0; c < C1; ++c) {
        float wc = adw_w[c * 9 + 4];           // center tap of 3x3
        float t1 = xmax[b * C1 + c] * wc + adw_b[c];
        float t2 = xavg[b * C1 + c] * wc + adw_b[c];
        acc += apw_w[o * C1 + c] * (t1 + t2);
    }
    g[idx] = sigm(acc);
}

// =====================================================================
// layernorm stats over channels per pixel (reads [n][c] layout)
// =====================================================================
__global__ void k_ln_stats(const float* __restrict__ dct, float* __restrict__ lmean,
                           float* __restrict__ lrstd) {
    int n = blockIdx.x;
    const float* p = dct + (size_t)n * CH;
    float v0 = p[threadIdx.x], v1 = p[threadIdx.x + 256];
    float s = v0 + v1, s2 = v0 * v0 + v1 * v1;
    __shared__ float sh[256], sh2[256];
    sh[threadIdx.x] = s; sh2[threadIdx.x] = s2; __syncthreads();
    for (int o = 128; o > 0; o >>= 1) {
        if (threadIdx.x < o) { sh[threadIdx.x] += sh[threadIdx.x + o]; sh2[threadIdx.x] += sh2[threadIdx.x + o]; }
        __syncthreads();
    }
    if (threadIdx.x == 0) {
        float mu = sh[0] * (1.0f / CH);
        float var = sh2[0] * (1.0f / CH) - mu * mu;
        lmean[n] = mu; lrstd[n] = rsqrtf(var + EPSN);
    }
}

// =====================================================================
// final fused epilogue: out = x * (d * sigmoid(LN(dc))) + x * g
// =====================================================================
__global__ void k_final(const float* __restrict__ x, const float* __restrict__ dcn,
                        const float* __restrict__ dccn, const float* __restrict__ lmean,
                        const float* __restrict__ lrstd, const float* __restrict__ ln_g,
                        const float* __restrict__ ln_b, const float* __restrict__ g,
                        float* __restrict__ out) {
    int idx = blockIdx.x * 256 + threadIdx.x;  // < 512*8192
    int c = idx >> 13, n = idx & 8191, b = n >> 12, hw = n & 4095;
    float dcv = dccn[idx];
    float lnv = (dcv - lmean[n]) * lrstd[n] * ln_g[c] + ln_b[c];
    float das = dcn[idx] * sigm(lnv);
    size_t xi = ((size_t)b * CH + c) * HW + hw;
    float xv = x[xi];
    out[xi] = xv * das + xv * g[(b << 9) | c];
}

// =====================================================================
// host
// =====================================================================
extern "C" void kernel_launch(void* const* d_in, const int* in_sizes, int n_in,
                              void* d_out, int out_size, void* d_ws, size_t ws_size,
                              hipStream_t stream) {
    const float* x      = (const float*)d_in[0];
    const float* dw_w   = (const float*)d_in[1];
    const float* dw_b   = (const float*)d_in[2];
    const float* pw_w   = (const float*)d_in[3];
    const float* pw_b   = (const float*)d_in[4];
    const float* off_w  = (const float*)d_in[5];
    const float* off_b  = (const float*)d_in[6];
    const float* dc_w   = (const float*)d_in[7];
    const float* dc_b   = (const float*)d_in[8];
    const float* ln_g   = (const float*)d_in[9];
    const float* ln_b   = (const float*)d_in[10];
    const float* adw_w  = (const float*)d_in[11];
    const float* adw_b  = (const float*)d_in[12];
    const float* apw_w  = (const float*)d_in[13];
    const float* apw_b  = (const float*)d_in[14];

    char* ws = (char*)d_ws;
    size_t cur = 0;
    auto take = [&](size_t bytes) { size_t o = cur; cur = (cur + bytes + 255) & ~(size_t)255; return o; };

    __bf16* d0b   = (__bf16*)(ws + take((size_t)C1 * NN * 2));     // dw output, bf16
    __bf16* wpwb  = (__bf16*)(ws + take((size_t)CH * C1 * 2));     // pw weight bf16
    __bf16* wdcb  = (__bf16*)(ws + take((size_t)CH * KDC * 2));    // dc weight bf16
    float*  dcn   = (float*) (ws + take((size_t)CH * NN * 4));     // d, [c][n]
    float*  imean = (float*) (ws + take(1024 * 4));
    float*  irstd = (float*) (ws + take(1024 * 4));
    float*  offm  = (float*) (ws + take((size_t)BB * 18 * HW * 4));
    __bf16* samp  = (__bf16*)(ws + take((size_t)KDC * NN * 2));    // samples bf16
    float*  dct   = (float*) (ws + take((size_t)NN * CH * 4));     // dc, [n][c]
    float*  dccn  = (float*) (ws + take((size_t)CH * NN * 4));     // dc, [c][n]
    float*  xmax  = (float*) (ws + take(512 * 4));
    float*  xavg  = (float*) (ws + take(512 * 4));
    float*  gbuf  = (float*) (ws + take(1024 * 4));
    float*  lmean = (float*) (ws + take((size_t)NN * 4));
    float*  lrstd = (float*) (ws + take((size_t)NN * 4));
    (void)ws_size; (void)in_sizes; (void)n_in; (void)out_size;

    dim3 blk(256);

    // weight conversions
    k_f32_to_bf16<<<(CH * C1 + 255) / 256, blk, 0, stream>>>(pw_w, wpwb, CH * C1);
    k_f32_to_bf16<<<(CH * KDC + 255) / 256, blk, 0, stream>>>(dc_w, wdcb, CH * KDC);

    // depthwise conv
    k_dwconv<<<(C1 * NN) / 256, blk, 0, stream>>>(x, dw_w, dw_b, d0b);

    // pointwise GEMM 512x256x8192 (WMMA bf16) -> dcn raw
    {
        dim3 grid(NN / 128, CH / 128);
        k_gemm_bf16<<<grid, blk, 0, stream>>>(wpwb, d0b, pw_b, dcn, nullptr, CH, NN, C1);
    }

    // instance norm
    k_inst_stats<<<1024, blk, 0, stream>>>(dcn, imean, irstd);
    k_inst_norm<<<(CH * NN) / 256, blk, 0, stream>>>(dcn, imean, irstd);

    // offset conv + sampling
    k_offconv<<<(18 * NN) / 256, blk, 0, stream>>>(dcn, off_w, off_b, offm);
    k_sample<<<((size_t)KDC * NN) / 256, blk, 0, stream>>>(dcn, offm, samp);

    // deform-conv GEMM 512x4608x8192 (WMMA bf16) -> dct [n][c] and dccn [c][n]
    {
        dim3 grid(NN / 128, CH / 128);
        k_gemm_bf16<<<grid, blk, 0, stream>>>(wdcb, samp, dc_b, dccn, dct, CH, NN, KDC);
    }

    // squeeze-excite branch
    k_x2_stats<<<512, blk, 0, stream>>>(x, xmax, xavg);
    k_gate<<<4, blk, 0, stream>>>(xmax, xavg, adw_w, adw_b, apw_w, apw_b, gbuf);

    // channel layernorm + fused epilogue
    k_ln_stats<<<NN, blk, 0, stream>>>(dct, lmean, lrstd);
    k_final<<<(CH * NN) / 256, blk, 0, stream>>>(x, dcn, dccn, lmean, lrstd,
                                                 ln_g, ln_b, gbuf, (float*)d_out);
}
